// MultiHeadAttention_23605140259316
// MI455X (gfx1250) — matrix operand order repaired and checked by reading, not compiled, not run on hardware
//
#include <hip/hip_runtime.h>
#include <cstdint>
#include <cstddef>

typedef __attribute__((ext_vector_type(16))) __bf16 v16bf;
typedef __attribute__((ext_vector_type(8)))  __bf16 v8bf;
typedef __attribute__((ext_vector_type(8)))  float  v8f;
typedef __attribute__((__vector_size__(16))) int    v4i;
typedef __attribute__((address_space(1)))    v4i    as1_v4i; // global
typedef __attribute__((address_space(3)))    v4i    as3_v4i; // LDS

union BF8U { v8bf v; unsigned int u[4]; __bf16 e[8]; };
union BF16V { v16bf v; v8bf p[2]; };

static constexpr int   B_  = 2;
static constexpr int   S_  = 2048;
static constexpr int   D_  = 2048;
static constexpr int   H_  = 16;
static constexpr int   DH_ = 128;
static constexpr int   BS_ = B_ * S_;
static constexpr float EPS_   = 1e-5f;
static constexpr float SCALE_ = 0.088388347648318447f; // 1/sqrt(DH)

#if __has_builtin(__builtin_amdgcn_cvt_pk_bf16_f32)
#define USE_PK_BF16 1
#else
#define USE_PK_BF16 0
#endif

#if __has_builtin(__builtin_amdgcn_mov_dpp8) && __has_builtin(__builtin_amdgcn_update_dpp)
#define USE_DPP_RED 1
#else
#define USE_DPP_RED 0
#endif

#if __has_builtin(__builtin_amdgcn_global_load_async_to_lds_b128) && \
    __has_builtin(__builtin_amdgcn_s_wait_asynccnt)
#define USE_ASYNC_LDS 1
#else
#define USE_ASYNC_LDS 0
#endif

__device__ __forceinline__ v8f wmma_bf16(v16bf a, v16bf b, v8f c) {
  return __builtin_amdgcn_wmma_f32_16x16x32_bf16(false, a, false, b,
                                                 (short)0, c, false, false);
}

// Convert+pack 8 floats -> 8 bf16 (v_cvt_pk_bf16_f32 when available).
__device__ __forceinline__ v8bf pack8(const float* f) {
  BF8U r;
#if USE_PK_BF16
#pragma unroll
  for (int j = 0; j < 4; ++j)
    r.u[j] = __builtin_bit_cast(unsigned int,
        __builtin_amdgcn_cvt_pk_bf16_f32(f[2 * j], f[2 * j + 1]));
#else
#pragma unroll
  for (int j = 0; j < 8; ++j) r.e[j] = (__bf16)f[j];
#endif
  return r.v;
}

// ---- cross-lane butterfly within each 16-lane half (rows of WMMA C tiles) ----
#if USE_DPP_RED
template <int M>
__device__ __forceinline__ float dpp8_xor(float v) {
  constexpr unsigned sel =
      ((0u ^ M) << 0)  | ((1u ^ M) << 3)  | ((2u ^ M) << 6)  | ((3u ^ M) << 9) |
      ((4u ^ M) << 12) | ((5u ^ M) << 15) | ((6u ^ M) << 18) | ((7u ^ M) << 21);
  return __builtin_bit_cast(float,
      __builtin_amdgcn_mov_dpp8(__builtin_bit_cast(int, v), sel));
}
__device__ __forceinline__ float dpp_ror8(float v) {
  // DPP16 row_ror:8 : within a 16-lane row, lane i gets lane (i+8)%16 == i^8
  return __builtin_bit_cast(float,
      __builtin_amdgcn_update_dpp(0, __builtin_bit_cast(int, v),
                                  0x128, 0xf, 0xf, true));
}
__device__ __forceinline__ float red16_max(float v) {
  v = fmaxf(v, dpp8_xor<1>(v));
  v = fmaxf(v, dpp8_xor<2>(v));
  v = fmaxf(v, dpp8_xor<4>(v));
  v = fmaxf(v, dpp_ror8(v));
  return v;
}
__device__ __forceinline__ float red16_sum(float v) {
  v += dpp8_xor<1>(v);
  v += dpp8_xor<2>(v);
  v += dpp8_xor<4>(v);
  v += dpp_ror8(v);
  return v;
}
#else
__device__ __forceinline__ float red16_max(float v) {
#pragma unroll
  for (int m = 8; m >= 1; m >>= 1) v = fmaxf(v, __shfl_xor(v, m, 32));
  return v;
}
__device__ __forceinline__ float red16_sum(float v) {
#pragma unroll
  for (int m = 8; m >= 1; m >>= 1) v += __shfl_xor(v, m, 32);
  return v;
}
#endif

// ---------------------------------------------------------------------------
// Tiled GEMM: out[M,N] = A[M,K] * W[K,N] + bias[N]
// 256 threads (8 waves), block tile 128x128, K-step 32.
// Double-buffered LDS in WMMA fragment layout; register prefetch 2 tiles ahead.
// ---------------------------------------------------------------------------
template <typename AT, typename OT>
__global__ __launch_bounds__(256) void gemm_wmma(
    const AT* __restrict__ A, const float* __restrict__ W,
    const float* __restrict__ bias, OT* __restrict__ out,
    int M, int N, int K)
{
  __shared__ __attribute__((aligned(32))) __bf16 sA[2][4096];
  __shared__ __attribute__((aligned(32))) __bf16 sB[2][4096];

  const int tid  = threadIdx.x;
  const int lane = tid & 31;
  const int wave = tid >> 5;
  const int wm   = wave & 3;
  const int wn   = wave >> 2;
  const int m0   = blockIdx.y * 128;
  const int n0   = blockIdx.x * 128;

  v8f acc[2][4] = {};

  const int am  = tid >> 1;        // A tile row 0..127
  const int aks = (tid & 1) << 4;  // A k-seg 0/16
  const int bn  = tid & 127;       // B tile col 0..127
  const int bkh = tid >> 7;        // B k-group 0/1

  float ra[16], rb[16];

  auto loadA = [&](int kk) {
    const AT* src = A + (size_t)(m0 + am) * K + kk + aks;
#pragma unroll
    for (int j = 0; j < 16; ++j) ra[j] = (float)src[j];
  };
  auto loadB = [&](int kk) {
    const float* src = W + (size_t)(kk + bkh * 16) * N + n0 + bn;
#pragma unroll
    for (int j = 0; j < 16; ++j) rb[j] = src[(size_t)j * N];
  };
  auto stage = [&](int buf) {
    { // A-fragment layout (lane = m + 16*(k/16), e = k%16)
      v8bf lo = pack8(ra), hi = pack8(ra + 8);
      const int sub = am >> 4, mrr = am & 15, eb = aks >> 1;
      *(v8bf*)&sA[buf][sub * 512 + (mrr + 16 * (aks >> 4)) * 16]     = lo;
      *(v8bf*)&sA[buf][sub * 512 + (mrr + 16 * (aks >> 4)) * 16 + 8] = hi;
    }
    { // B-fragment layout (lane = n + 16*(k/16), e = k%16)
      v8bf lo = pack8(rb), hi = pack8(rb + 8);
      const int sub = bn >> 4, nr = bn & 15;
      const int off = sub * 512 + (nr + 16 * bkh) * 16;
      *(v8bf*)&sB[buf][off]     = lo;
      *(v8bf*)&sB[buf][off + 8] = hi;
    }
  };

  const int T = K >> 5;
  loadA(0); loadB(0);
  stage(0);
  if (T > 1) { loadA(32); loadB(32); }

  for (int t = 0; t < T; ++t) {
    __syncthreads();
    if (t + 1 < T) stage((t + 1) & 1);
    if (t + 2 < T) { loadA((t + 2) << 5); loadB((t + 2) << 5); }

    const __bf16* cA = &sA[t & 1][0];
    const __bf16* cB = &sB[t & 1][0];
    v16bf a0 = *(const v16bf*)&cA[(2 * wm + 0) * 512 + lane * 16];
    v16bf a1 = *(const v16bf*)&cA[(2 * wm + 1) * 512 + lane * 16];
#pragma unroll
    for (int j = 0; j < 4; ++j) {
      v16bf bf = *(const v16bf*)&cB[(4 * wn + j) * 512 + lane * 16];
      acc[0][j] = wmma_bf16(a0, bf, acc[0][j]);
      acc[1][j] = wmma_bf16(a1, bf, acc[1][j]);
    }
  }

  const int mr = (lane >> 4) << 3;
  const int nc = lane & 15;
#pragma unroll
  for (int i = 0; i < 2; ++i) {
#pragma unroll
    for (int j = 0; j < 4; ++j) {
      const int gm = m0 + wm * 32 + i * 16 + mr;
      const int gn = n0 + wn * 64 + j * 16 + nc;
      const float bv = bias[gn];
#pragma unroll
      for (int r = 0; r < 8; ++r)
        out[(size_t)(gm + r) * N + gn] = (OT)(acc[i][j][r] + bv);
    }
  }
}

// ---------------------------------------------------------------------------
// Flash attention. Grid (S/128, H, B), 256 threads (8 waves).
// K tile staged global->LDS with ASYNC loads (fragment-swizzled dest) when
// available; V tile via VGPR path. O written [B,H,S,DH]-flat (reference's
// buggy reshape semantics).
// ---------------------------------------------------------------------------
__global__ __launch_bounds__(256) void flash_attn(
    const __bf16* __restrict__ Qb, const __bf16* __restrict__ Kb,
    const __bf16* __restrict__ Vb, float* __restrict__ O)
{
  __shared__ __attribute__((aligned(32))) __bf16 sK[4096];
  __shared__ __attribute__((aligned(32))) __bf16 sV[4096];
  __shared__ __attribute__((aligned(32))) __bf16 sP[4096];

  const int tid = threadIdx.x, lane = tid & 31, wave = tid >> 5;
  const int qt = blockIdx.x, h = blockIdx.y, b = blockIdx.z;
  const int qrow0 = qt * 128 + wave * 16;
  const int ml = lane & 15, hf = lane >> 4;

  v16bf qf[4];
  {
    const __bf16* qs = Qb + ((size_t)b * S_ + qrow0 + ml) * D_ + h * DH_;
#pragma unroll
    for (int f = 0; f < 4; ++f) {
      BF16V u;
      u.p[0] = *(const v8bf*)(qs + f * 32 + 16 * hf);
      u.p[1] = *(const v8bf*)(qs + f * 32 + 16 * hf + 8);
      qf[f] = u.v;
    }
  }

  v8f o[8] = {};
  float mrow[8], lrow[8];
#pragma unroll
  for (int r = 0; r < 8; ++r) { mrow[r] = -3.0e38f; lrow[r] = 0.f; }

  const int kseq = tid >> 3;        // 0..31
  const int kdhs = (tid & 7) << 4;  // 0..112
  const int vdh  = tid & 127;       // 0..127
  const int vsh  = tid >> 7;        // 0/1

  for (int kt = 0; kt < S_; kt += 32) {
    { // K tile -> B-fragment layout of K^T (k-dim = dh, n = seq)
      const __bf16* src = Kb + ((size_t)b * S_ + kt + kseq) * D_ + h * DH_ + kdhs;
      const int kstep = kdhs >> 5;
      const int klq   = (kdhs >> 4) & 1;
      const int off = (kstep * 2 + (kseq >> 4)) * 512 + ((kseq & 15) + 16 * klq) * 16;
#if USE_ASYNC_LDS
      __builtin_amdgcn_global_load_async_to_lds_b128(
          (as1_v4i*)src, (as3_v4i*)&sK[off], 0, 0);
      __builtin_amdgcn_global_load_async_to_lds_b128(
          (as1_v4i*)(src + 8), (as3_v4i*)&sK[off + 8], 0, 0);
#else
      v8bf k0 = *(const v8bf*)src;
      v8bf k1 = *(const v8bf*)(src + 8);
      *(v8bf*)&sK[off]     = k0;
      *(v8bf*)&sK[off + 8] = k1;
#endif
      if (kt + 32 < S_) __builtin_prefetch(src + (size_t)32 * D_, 0, 1);
    }
    { // V tile -> B-fragment layout (k = seq, n = dh)
      const __bf16* src = Vb + ((size_t)b * S_ + kt + vsh * 16) * D_ + h * DH_ + vdh;
      BF8U t0, t1;
#pragma unroll
      for (int j = 0; j < 8; ++j) {
        t0.e[j] = src[(size_t)j * D_];
        t1.e[j] = src[(size_t)(j + 8) * D_];
      }
      const int off = (vdh >> 4) * 512 + ((vdh & 15) + 16 * vsh) * 16;
      *(v8bf*)&sV[off]     = t0.v;
      *(v8bf*)&sV[off + 8] = t1.v;
      if (kt + 32 < S_) __builtin_prefetch(src + (size_t)32 * D_, 0, 1);
    }
#if USE_ASYNC_LDS
    __builtin_amdgcn_s_wait_asynccnt(0);
#endif
    __syncthreads();

    // S strip (16x32) = Q(16x128) @ K^T over 4 WMMA K-steps
    v8f s0 = {}, s1 = {};
#pragma unroll
    for (int ks = 0; ks < 4; ++ks) {
      v16bf b0 = *(const v16bf*)&sK[(ks * 2 + 0) * 512 + lane * 16];
      v16bf b1 = *(const v16bf*)&sK[(ks * 2 + 1) * 512 + lane * 16];
      s0 = wmma_bf16(qf[ks], b0, s0);
      s1 = wmma_bf16(qf[ks], b1, s1);
    }

    // Online softmax: row m = r + 8*hf lives on one 16-lane half (DPP butterflies).
    float corr[8];
#pragma unroll
    for (int r = 0; r < 8; ++r) {
      float a0 = s0[r] * SCALE_;
      float a1 = s1[r] * SCALE_;
      const float mx   = red16_max(fmaxf(a0, a1));
      const float mnew = fmaxf(mrow[r], mx);
      const float c  = __expf(mrow[r] - mnew);
      const float p0 = __expf(a0 - mnew);
      const float p1 = __expf(a1 - mnew);
      const float rs = red16_sum(p0 + p1);
      lrow[r] = lrow[r] * c + rs;
      mrow[r] = mnew;
      corr[r] = c;
      s0[r] = p0;
      s1[r] = p1;
    }
#pragma unroll
    for (int nt = 0; nt < 8; ++nt)
#pragma unroll
      for (int r = 0; r < 8; ++r) o[nt][r] *= corr[r];

    // Stage P (16x32 bf16) into per-wave A-fragment layout.
    const int pb = wave * 512;
#pragma unroll
    for (int r = 0; r < 8; ++r) {
      const int mm = r + 8 * hf;
#pragma unroll
      for (int t = 0; t < 2; ++t) {
        const int k  = ml + 16 * t;
        const int e  = k & 15;
        const int hk = k >> 4;
        sP[pb + (mm + 16 * hk) * 16 + e] = (__bf16)(t ? s1[r] : s0[r]);
      }
    }
    __syncthreads();

    v16bf pf = *(const v16bf*)&sP[pb + lane * 16];
#pragma unroll
    for (int nt = 0; nt < 8; ++nt) {
      v16bf vb = *(const v16bf*)&sV[nt * 512 + lane * 16];
      o[nt] = wmma_bf16(pf, vb, o[nt]);
    }
    __syncthreads();
  }

  float inv[8];
#pragma unroll
  for (int r = 0; r < 8; ++r) inv[r] = 1.f / lrow[r];
#pragma unroll
  for (int nt = 0; nt < 8; ++nt) {
#pragma unroll
    for (int r = 0; r < 8; ++r) {
      const int mm  = r + 8 * hf;
      const int col = nt * 16 + ml;
      O[(((size_t)b * H_ + h) * S_ + qrow0 + mm) * DH_ + col] = o[nt][r] * inv[r];
    }
  }
}

// ---------------------------------------------------------------------------
// Fused residual add + LayerNorm over D. One block per row, 256 threads.
// ---------------------------------------------------------------------------
template <typename OT>
__global__ __launch_bounds__(256) void resid_ln(
    const float* __restrict__ X, const float* __restrict__ R,
    const float* __restrict__ gamma, const float* __restrict__ beta,
    OT* __restrict__ out)
{
  const int row = blockIdx.x;
  const int tid = threadIdx.x, lane = tid & 31, wave = tid >> 5;
  const float* a = X + (size_t)row * D_;
  const float* c = R + (size_t)row * D_;
  float v[8];
  float s = 0.f, ss = 0.f;
#pragma unroll
  for (int j = 0; j < 8; ++j) {
    const int i = tid + j * 256;
    const float t = a[i] + c[i];
    v[j] = t; s += t; ss += t * t;
  }
#pragma unroll
  for (int msk = 16; msk >= 1; msk >>= 1) {
    s  += __shfl_xor(s,  msk, 32);
    ss += __shfl_xor(ss, msk, 32);
  }
  __shared__ float rs[8], rss[8];
  if (lane == 0) { rs[wave] = s; rss[wave] = ss; }
  __syncthreads();
  s = 0.f; ss = 0.f;
#pragma unroll
  for (int w = 0; w < 8; ++w) { s += rs[w]; ss += rss[w]; }
  const float mu   = s * (1.f / D_);
  const float var  = ss * (1.f / D_) - mu * mu;
  const float rstd = rsqrtf(var + EPS_);
#pragma unroll
  for (int j = 0; j < 8; ++j) {
    const int i = tid + j * 256;
    out[(size_t)row * D_ + i] = (OT)((v[j] - mu) * rstd * gamma[i] + beta[i]);
  }
}

// ---------------------------------------------------------------------------
extern "C" void kernel_launch(void* const* d_in, const int* in_sizes, int n_in,
                              void* d_out, int out_size, void* d_ws, size_t ws_size,
                              hipStream_t stream) {
  const float* x     = (const float*)d_in[0];
  const float* wq    = (const float*)d_in[1];
  const float* bq    = (const float*)d_in[2];
  const float* wk    = (const float*)d_in[3];
  const float* bk    = (const float*)d_in[4];
  const float* wv    = (const float*)d_in[5];
  const float* bv    = (const float*)d_in[6];
  const float* wf    = (const float*)d_in[7];
  const float* bfv   = (const float*)d_in[8];
  const float* gamma = (const float*)d_in[9];
  const float* beta  = (const float*)d_in[10];

  char* ws = (char*)d_ws;
  const size_t MB = 1024ull * 1024ull;
  __bf16* Qb  = (__bf16*)(ws + 0);        // 16MB
  __bf16* Kbf = (__bf16*)(ws + 16 * MB);  // 16MB
  __bf16* Vbf = (__bf16*)(ws + 32 * MB);  // 16MB
  float*  O   = (float*)(ws + 48 * MB);   // 32MB
  __bf16* H1  = (__bf16*)(ws + 0);        // reuses Qb after attention
  float*  H2  = (float*)(ws + 16 * MB);   // reuses Kb+Vb after attention

  dim3 blk(256);
  dim3 gG(D_ / 128, BS_ / 128);

  gemm_wmma<float, __bf16><<<gG, blk, 0, stream>>>(x, wq, bq, Qb,  BS_, D_, D_);
  gemm_wmma<float, __bf16><<<gG, blk, 0, stream>>>(x, wk, bk, Kbf, BS_, D_, D_);
  gemm_wmma<float, __bf16><<<gG, blk, 0, stream>>>(x, wv, bv, Vbf, BS_, D_, D_);

  flash_attn<<<dim3(S_ / 128, H_, B_), blk, 0, stream>>>(Qb, Kbf, Vbf, O);

  resid_ln<__bf16><<<BS_, blk, 0, stream>>>(O, x, gamma, beta, H1);

  gemm_wmma<__bf16, float><<<gG, blk, 0, stream>>>(H1, wf, bfv, H2, BS_, D_, D_);

  resid_ln<float><<<BS_, blk, 0, stream>>>(H2, O, gamma, beta, (float*)d_out);
}
